// NetGIN_53549652247067
// MI455X (gfx1250) — matrix operand order repaired and checked by reading, not compiled, not run on hardware
//
#include <hip/hip_runtime.h>
#include <hip/hip_bf16.h>
#include <cstdint>

// ---------------- problem constants (match reference) ----------------
#define NNODES   100000
#define NALLN    100000
#define NEDGE    1600000
#define BGRAPH   512
#define FIN0     242
#define FALLD    652
#define DIMD     128
#define BN_EPS_F 1e-5f

// ---------------- vector types ----------------
typedef __bf16  bf16_t;
typedef float   f2v  __attribute__((ext_vector_type(2)));
typedef float   f4v  __attribute__((ext_vector_type(4)));
typedef float   v8f  __attribute__((ext_vector_type(8)));
typedef __bf16  v16bf __attribute__((ext_vector_type(16)));

union Frag32B { f4v f[2]; v16bf v; };

// =====================================================================
// bf16 GEMM, dout fixed = 128:  C[M,128] = A[M,Kpad] @ W[Kreal,128]
// A: bf16 row-major (zero padded to Kpad), W: fp32 row-major [Kreal,128].
// 256 threads = 8 waves; each wave owns a 32-row strip (two 16-row A
// fragments sharing every B fragment -> 2 WMMAs per LDS B read) across
// all 8 column tiles. W is staged to LDS transposed as bf16 with a
// 136-element pitch (68 dwords -> conflict-free ds_load_b128).
// Requires M % 32 == 0.
// =====================================================================
__global__ __launch_bounds__(256)
void gemm_bf16_n128(const bf16_t* __restrict__ A, int lda,
                    const float* __restrict__ W, int Kreal,
                    const float* __restrict__ bias,
                    void* __restrict__ Cout, int ldc, int col_off,
                    int M, int Kpad, int relu, int out_bf16)
{
    __shared__ bf16_t WT[128 * 136];   // [dout=128][Kchunk=128 (+8 pad)]

    const int tid  = threadIdx.x;
    const int wave = tid >> 5;
    const int lane = tid & 31;
    const int lo   = lane & 15;
    const int hi   = lane >> 4;
    const int strip = blockIdx.x * 8 + wave;
    const int m0    = strip * 32;
    const bool valid = (m0 < M);
    const int arow0 = m0 + lo;
    const int arow1 = m0 + 16 + lo;

    v8f acc0[8], acc1[8];
    #pragma unroll
    for (int i = 0; i < 8; ++i)
        #pragma unroll
        for (int r = 0; r < 8; ++r) { acc0[i][r] = 0.0f; acc1[i][r] = 0.0f; }

    for (int kc = 0; kc < Kpad; kc += 128) {
        int kchunk = Kpad - kc; if (kchunk > 128) kchunk = 128;

        // stage W^T chunk into LDS as bf16 (coalesced over n)
        for (int idx = tid; idx < (kchunk << 7); idx += 256) {
            int k = idx >> 7;
            int n = idx & 127;
            int kg = kc + k;
            float wv = (kg < Kreal) ? W[(size_t)kg * 128 + n] : 0.0f;
            WT[n * 136 + k] = (bf16_t)wv;
        }
        __syncthreads();

        if (valid) {
            for (int k0 = 0; k0 < kchunk; k0 += 32) {
                // A fragments (16-bit A layout: lanes0-15 K0-7/16-23,
                // lanes16-31 K8-15/24-31)
                const bf16_t* ap0 = A + (size_t)arow0 * lda + (kc + k0) + hi * 8;
                const bf16_t* ap1 = A + (size_t)arow1 * lda + (kc + k0) + hi * 8;
                Frag32B a0, a1;
                a0.f[0] = *(const f4v*)(ap0);
                a0.f[1] = *(const f4v*)(ap0 + 16);
                a1.f[0] = *(const f4v*)(ap1);
                a1.f[1] = *(const f4v*)(ap1 + 16);
                #pragma unroll
                for (int nt = 0; nt < 8; ++nt) {
                    // B fragment: column (nt*16+lo), K k0+hi*8 .. +7 and k0+16+hi*8 .. +7
                    const bf16_t* bp = &WT[(nt * 16 + lo) * 136 + k0 + hi * 8];
                    Frag32B bu;
                    bu.f[0] = *(const f4v*)(bp);
                    bu.f[1] = *(const f4v*)(bp + 16);
                    acc0[nt] = __builtin_amdgcn_wmma_f32_16x16x32_bf16(
                        false, a0.v, false, bu.v, (short)0, acc0[nt], false, false);
                    acc1[nt] = __builtin_amdgcn_wmma_f32_16x16x32_bf16(
                        false, a1.v, false, bu.v, (short)0, acc1[nt], false, false);
                }
            }
        }
        __syncthreads();
    }

    if (valid) {
        // C/D layout: VGPR r -> row r (lanes 0-15), row r+8 (lanes 16-31)
        const int rbase0 = m0 + hi * 8;
        const int rbase1 = m0 + 16 + hi * 8;
        #pragma unroll
        for (int nt = 0; nt < 8; ++nt) {
            int col = nt * 16 + lo;
            float bv = bias ? bias[col] : 0.0f;
            #pragma unroll
            for (int r = 0; r < 8; ++r) {
                float v0 = acc0[nt][r] + bv;
                float v1 = acc1[nt][r] + bv;
                if (relu) { v0 = fmaxf(v0, 0.0f); v1 = fmaxf(v1, 0.0f); }
                size_t off0 = (size_t)(rbase0 + r) * ldc + col_off + col;
                size_t off1 = (size_t)(rbase1 + r) * ldc + col_off + col;
                if (out_bf16) {
                    ((bf16_t*)Cout)[off0] = (bf16_t)v0;
                    ((bf16_t*)Cout)[off1] = (bf16_t)v1;
                } else {
                    ((float*)Cout)[off0] = v0;
                    ((float*)Cout)[off1] = v1;
                }
            }
        }
    }
}

// ---------------- scatter-add: agg[dst] += h[src] (VEC = 2 or 4) ----------------
template <int VEC>
__global__ void scatter_add_kernel(const float* __restrict__ h, int ldh,
                                   const int* __restrict__ src,
                                   const int* __restrict__ dst,
                                   int ne, int dvec,
                                   float* __restrict__ agg, int lda)
{
    size_t idx = (size_t)blockIdx.x * blockDim.x + threadIdx.x;
    size_t total = (size_t)ne * dvec;
    if (idx >= total) return;
    int e = (int)(idx / dvec);
    int f = (int)(idx % dvec) * VEC;
    int s = src[e], d = dst[e];
    const float* hp = h + (size_t)s * ldh + f;
    float*       ap = agg + (size_t)d * lda + f;
    if (VEC == 4) {
        f4v v = *(const f4v*)hp;
        atomicAdd(ap + 0, v.x);
        atomicAdd(ap + 1, v.y);
        atomicAdd(ap + 2, v.z);
        atomicAdd(ap + 3, v.w);
    } else {
        f2v v = *(const f2v*)hp;
        atomicAdd(ap + 0, v.x);
        atomicAdd(ap + 1, v.y);
    }
}

// ---------------- z0 = (1+eps)*h + agg, store bf16 (zero padded) ----------------
__global__ void make_z0_kernel(const float* __restrict__ h, int ldh,
                               const float* __restrict__ agg, int lda,
                               const float* __restrict__ eps_p,
                               bf16_t* __restrict__ z0, int ldz,
                               int n, int fin)
{
    size_t idx = (size_t)blockIdx.x * blockDim.x + threadIdx.x;
    size_t total = (size_t)n * ldz;
    if (idx >= total) return;
    int i = (int)(idx / ldz);
    int f = (int)(idx % ldz);
    float val = 0.0f;
    if (f < fin) {
        float e = 1.0f + eps_p[0];
        val = e * h[(size_t)i * ldh + f] + agg[(size_t)i * lda + f];
    }
    z0[idx] = (bf16_t)val;
}

// ---------------- batchnorm stats: per-feature sum / sumsq ----------------
__global__ void bn_stats_kernel(const float* __restrict__ v, int n,
                                float* __restrict__ sum, float* __restrict__ sq)
{
    int f  = threadIdx.x;              // blockDim.x == 128
    int r0 = blockIdx.x * 512;
    int r1 = r0 + 512; if (r1 > n) r1 = n;
    float s = 0.0f, q = 0.0f;
    for (int r = r0; r < r1; ++r) {
        float x = v[(size_t)r * 128 + f];
        s += x; q += x * x;
    }
    atomicAdd(&sum[f], s);
    atomicAdd(&sq[f], q);
}

__global__ void bn_apply_kernel(const float* __restrict__ v,
                                const float* __restrict__ sum,
                                const float* __restrict__ sq,
                                const float* __restrict__ gamma,
                                const float* __restrict__ beta,
                                float* __restrict__ out, int ldo, int n)
{
    size_t idx = (size_t)blockIdx.x * blockDim.x + threadIdx.x;
    size_t total = (size_t)n * 128;
    if (idx >= total) return;
    int i = (int)(idx / 128);
    int f = (int)(idx & 127);
    float m   = sum[f] / (float)n;
    float var = sq[f] / (float)n - m * m;
    out[(size_t)i * ldo + f] = (v[idx] - m) * rsqrtf(var + BN_EPS_F) * gamma[f] + beta[f];
}

// ---------------- segment mean-pool pieces ----------------
__global__ void pool_add_kernel(const float* __restrict__ in, int ldin,
                                const int* __restrict__ seg, int n, int F,
                                float* __restrict__ out)
{
    size_t idx = (size_t)blockIdx.x * blockDim.x + threadIdx.x;
    size_t total = (size_t)n * F;
    if (idx >= total) return;
    int i = (int)(idx / F);
    int f = (int)(idx % F);
    atomicAdd(&out[(size_t)seg[i] * F + f], in[(size_t)i * ldin + f]);
}

__global__ void count_kernel(const int* __restrict__ seg, int n, float* __restrict__ cnt)
{
    int i = blockIdx.x * blockDim.x + threadIdx.x;
    if (i < n) atomicAdd(&cnt[seg[i]], 1.0f);
}

// ---------------- concat pooled features -> bf16, padded to 1184 ----------------
__global__ void make_gcat_kernel(const float* __restrict__ g,
                                 const float* __restrict__ gall,
                                 const float* __restrict__ cntN,
                                 const float* __restrict__ cntA,
                                 bf16_t* __restrict__ out)
{
    int idx = blockIdx.x * blockDim.x + threadIdx.x;
    if (idx >= BGRAPH * 1184) return;
    int b = idx / 1184;
    int j = idx % 1184;
    float val = 0.0f;
    if (j < 512)       val = g[b * 512 + j]            / fmaxf(cntN[b], 1.0f);
    else if (j < 1164) val = gall[b * FALLD + (j-512)] / fmaxf(cntA[b], 1.0f);
    out[idx] = (bf16_t)val;
}

// ---------------- final FC: [512,128] @ [128,1] + b ----------------
__global__ void fc4_kernel(const bf16_t* __restrict__ in,
                           const float* __restrict__ W,
                           const float* __restrict__ bias,
                           float* __restrict__ out)
{
    int b = blockIdx.x * blockDim.x + threadIdx.x;
    if (b >= BGRAPH) return;
    float s = bias[0];
    #pragma unroll 8
    for (int j = 0; j < 128; ++j) s += (float)in[b * 128 + j] * W[j];
    out[b] = s;
}

// =====================================================================
// host driver
// =====================================================================
static inline unsigned gridFor(size_t total, unsigned tpb) {
    return (unsigned)((total + tpb - 1) / tpb);
}

extern "C" void kernel_launch(void* const* d_in, const int* in_sizes, int n_in,
                              void* d_out, int out_size, void* d_ws, size_t ws_size,
                              hipStream_t stream)
{
    (void)in_sizes; (void)n_in; (void)out_size; (void)ws_size;

    const float* x         = (const float*)d_in[0];
    const float* x_all     = (const float*)d_in[1];
    const int*   ei1       = (const int*)d_in[2];   // [2,E] row-major: src then dst
    const int*   ei2       = (const int*)d_in[3];
    const int*   batch     = (const int*)d_in[4];
    const int*   batch_all = (const int*)d_in[5];
    // conv_params: 4 layers x 2 branches x (eps, W1, b1, W2, b2) -> 40 arrays at [6..45]
    auto cp = [&](int k, int br, int j) { return (const float*)d_in[6 + k*10 + br*5 + j]; };
    // mlp_params: 4 x (Wa, ba, Wb, bb) at [46..61]
    auto mp = [&](int k, int j) { return (const float*)d_in[46 + k*4 + j]; };
    // bn_params: 4 x (gamma, beta) at [62..69]
    auto bnp = [&](int k, int j) { return (const float*)d_in[62 + k*2 + j]; };
    // fc_params: 4 x (W, b) at [70..77]
    auto fcp = [&](int i, int j) { return (const float*)d_in[70 + i*2 + j]; };

    // -------- workspace carve --------
    char* w = (char*)d_ws;
    auto carve = [&](size_t bytes) { char* p = w; w += (bytes + 255) & ~(size_t)255; return p; };
    float*  agg   = (float*) carve((size_t)NNODES * 256 * 4);   // conv aggregate
    bf16_t* z0b   = (bf16_t*)carve((size_t)NNODES * 256 * 2);   // (1+eps)h+agg in bf16
    bf16_t* tb    = (bf16_t*)carve((size_t)NNODES * 128 * 2);   // relu(z0@W1+b1)
    bf16_t* hcatb = (bf16_t*)carve((size_t)NNODES * 256 * 2);   // [h1|h2] bf16
    bf16_t* ub    = (bf16_t*)carve((size_t)NNODES * 128 * 2);   // relu(hcat@Wa+ba)
    float*  vbuf  = (float*) carve((size_t)NNODES * 128 * 4);   // u@Wb+bb (pre-BN)
    float*  LO    = (float*) carve((size_t)NNODES * 512 * 4);   // concat of 4 BN outputs
    float*  bnsum = (float*) carve(128 * 4);
    float*  bnsq  = (float*) carve(128 * 4);
    float*  gpool = (float*) carve((size_t)BGRAPH * 512 * 4);
    float*  gall  = (float*) carve((size_t)BGRAPH * FALLD * 4);
    float*  cntN  = (float*) carve(BGRAPH * 4);
    float*  cntA  = (float*) carve(BGRAPH * 4);
    bf16_t* gcatb = (bf16_t*)carve((size_t)BGRAPH * 1184 * 2);
    bf16_t* f1b   = (bf16_t*)carve((size_t)BGRAPH * 128 * 2);
    bf16_t* f2b   = (bf16_t*)carve((size_t)BGRAPH * 128 * 2);
    bf16_t* f3b   = (bf16_t*)carve((size_t)BGRAPH * 128 * 2);

    const unsigned TPB = 256;
    // 32 rows per wave, 8 waves per block
    const unsigned gemmBlocksN = (NNODES / 32 + 7) / 8;   // 3125 strips -> 391 blocks
    const unsigned gemmBlocksB = (BGRAPH / 32 + 7) / 8;   // 16 strips -> 2 blocks

    // ================= 4 GIN layers =================
    for (int k = 0; k < 4; ++k) {
        const float* hin; int ldh, fin, Kpad;
        if (k == 0) { hin = x;                ldh = FIN0; fin = FIN0; Kpad = 256; }
        else        { hin = LO + (k-1)*128;   ldh = 512;  fin = 128;  Kpad = 128; }

        for (int br = 0; br < 2; ++br) {
            const int* ei  = br ? ei2 : ei1;
            const int* src = ei;            // edge_index[0]
            const int* dst = ei + NEDGE;    // edge_index[1]

            hipMemsetAsync(agg, 0, (size_t)NNODES * Kpad * 4, stream);

            if (fin == 128) {
                int dvec = 128 / 4;
                size_t sc_total = (size_t)NEDGE * dvec;
                scatter_add_kernel<4><<<gridFor(sc_total, TPB), TPB, 0, stream>>>(
                    hin, ldh, src, dst, NEDGE, dvec, agg, Kpad);
            } else {
                int dvec = fin / 2;
                size_t sc_total = (size_t)NEDGE * dvec;
                scatter_add_kernel<2><<<gridFor(sc_total, TPB), TPB, 0, stream>>>(
                    hin, ldh, src, dst, NEDGE, dvec, agg, Kpad);
            }

            size_t z0_total = (size_t)NNODES * Kpad;
            make_z0_kernel<<<gridFor(z0_total, TPB), TPB, 0, stream>>>(
                hin, ldh, agg, Kpad, cp(k, br, 0), z0b, Kpad, NNODES, fin);

            // t = relu(z0 @ W1 + b1) -> bf16 [N,128]
            gemm_bf16_n128<<<gemmBlocksN, TPB, 0, stream>>>(
                z0b, Kpad, cp(k, br, 1), fin, cp(k, br, 2),
                tb, 128, 0, NNODES, Kpad, /*relu=*/1, /*bf16=*/1);

            // h_br = relu(t @ W2 + b2) -> bf16 into hcat half
            gemm_bf16_n128<<<gemmBlocksN, TPB, 0, stream>>>(
                tb, 128, cp(k, br, 3), 128, cp(k, br, 4),
                hcatb, 256, br * 128, NNODES, 128, /*relu=*/1, /*bf16=*/1);
        }

        // u = relu(hcat @ Wa + ba) -> bf16 [N,128]
        gemm_bf16_n128<<<gemmBlocksN, TPB, 0, stream>>>(
            hcatb, 256, mp(k, 0), 256, mp(k, 1),
            ub, 128, 0, NNODES, 256, 1, 1);

        // v = u @ Wb + bb -> fp32 [N,128]
        gemm_bf16_n128<<<gemmBlocksN, TPB, 0, stream>>>(
            ub, 128, mp(k, 2), 128, mp(k, 3),
            vbuf, 128, 0, NNODES, 128, 0, 0);

        // BN (training mode, biased var) -> LO[:, k*128 : k*128+128]
        hipMemsetAsync(bnsum, 0, 128 * 4, stream);
        hipMemsetAsync(bnsq,  0, 128 * 4, stream);
        bn_stats_kernel<<<gridFor(NNODES, 512), 128, 0, stream>>>(vbuf, NNODES, bnsum, bnsq);
        bn_apply_kernel<<<gridFor((size_t)NNODES * 128, TPB), TPB, 0, stream>>>(
            vbuf, bnsum, bnsq, bnp(k, 0), bnp(k, 1), LO + k * 128, 512, NNODES);
    }

    // ================= pooling =================
    hipMemsetAsync(gpool, 0, (size_t)BGRAPH * 512 * 4, stream);
    hipMemsetAsync(gall,  0, (size_t)BGRAPH * FALLD * 4, stream);
    hipMemsetAsync(cntN,  0, BGRAPH * 4, stream);
    hipMemsetAsync(cntA,  0, BGRAPH * 4, stream);

    pool_add_kernel<<<gridFor((size_t)NNODES * 512, TPB), TPB, 0, stream>>>(
        LO, 512, batch, NNODES, 512, gpool);
    pool_add_kernel<<<gridFor((size_t)NALLN * FALLD, TPB), TPB, 0, stream>>>(
        x_all, FALLD, batch_all, NALLN, FALLD, gall);
    count_kernel<<<gridFor(NNODES, TPB), TPB, 0, stream>>>(batch, NNODES, cntN);
    count_kernel<<<gridFor(NALLN, TPB), TPB, 0, stream>>>(batch_all, NALLN, cntA);

    make_gcat_kernel<<<gridFor((size_t)BGRAPH * 1184, TPB), TPB, 0, stream>>>(
        gpool, gall, cntN, cntA, gcatb);

    // ================= FC head =================
    gemm_bf16_n128<<<gemmBlocksB, TPB, 0, stream>>>(
        gcatb, 1184, fcp(0, 0), 1164, fcp(0, 1), f1b, 128, 0, BGRAPH, 1184, 1, 1);
    gemm_bf16_n128<<<gemmBlocksB, TPB, 0, stream>>>(
        f1b, 128, fcp(1, 0), 128, fcp(1, 1), f2b, 128, 0, BGRAPH, 128, 1, 1);
    gemm_bf16_n128<<<gemmBlocksB, TPB, 0, stream>>>(
        f2b, 128, fcp(2, 0), 128, fcp(2, 1), f3b, 128, 0, BGRAPH, 128, 1, 1);
    fc4_kernel<<<gridFor(BGRAPH, TPB), TPB, 0, stream>>>(
        f3b, fcp(3, 0), fcp(3, 1), (float*)d_out);
}
